// UltraLight_VM_UNet_70875550319167
// MI455X (gfx1250) — compile-verified, matrix operand order checked by reading, not run on hardware
//
#include <hip/hip_runtime.h>

// ---------------------------------------------------------------------------
// UltraLight-VM-UNet forward for gfx1250 (MI455X).
// Network is bandwidth-bound (everything fits in the 192MB L2 mid-net), so
// kernels are fused memory-pass kernels; GEMM-shaped work (Mamba/PVM
// projections) uses v_wmma_f32_16x16x32_f16 with pre-packed f16 fragment
// buffers so the WMMA loop is pure b128 loads + wmma (no exec-mask
// predication, no per-element waits).
// ---------------------------------------------------------------------------

typedef _Float16 f16_t;
typedef __attribute__((ext_vector_type(16))) _Float16 v16h;
typedef __attribute__((ext_vector_type(8)))  float    v8f;

#define DEVFN static __device__ __forceinline__

DEVFN float geluf(float x)    { return 0.5f * x * (1.0f + erff(x * 0.70710678118654752f)); }
DEVFN float sigmf(float x)    { return 1.0f / (1.0f + __expf(-x)); }
DEVFN float siluf(float x)    { return x * sigmf(x); }
DEVFN float softplusf_(float x){ return (x > 20.0f) ? x : log1pf(__expf(x)); }

// ---------------------------------------------------------------------------
// Generic direct conv2d (NCHW), bias, padding, dilation. fp32 VALU: the convs
// here have Cin=2..24 -> tensor-op-starved as GEMM; bandwidth bound anyway.
// ---------------------------------------------------------------------------
__global__ __launch_bounds__(256) void k_conv2d(
    const float* __restrict__ in, const float* __restrict__ w,
    const float* __restrict__ bias, float* __restrict__ out,
    int B, int Ci, int Co, int H, int W, int K, int pad, int dil)
{
  int i = blockIdx.x * blockDim.x + threadIdx.x;
  int total = B * Co * H * W;
  if (i >= total) return;
  int x = i % W, y = (i / W) % H, co = (i / (W * H)) % Co, b = i / (W * H * Co);
  float acc = bias ? bias[co] : 0.0f;
  for (int ci = 0; ci < Ci; ++ci)
    for (int ky = 0; ky < K; ++ky) {
      int yy = y - pad + ky * dil;
      if (yy < 0 || yy >= H) continue;
      for (int kx = 0; kx < K; ++kx) {
        int xx = x - pad + kx * dil;
        if (xx < 0 || xx >= W) continue;
        acc += in[((b * Ci + ci) * H + yy) * W + xx] * w[((co * Ci + ci) * K + ky) * K + kx];
      }
    }
  out[i] = acc;
}

// ---------------------------------------------------------------------------
// GroupNorm (4 groups), in-place. One block per (batch, group), LDS reduce.
// ---------------------------------------------------------------------------
__global__ __launch_bounds__(256) void k_gn(
    float* __restrict__ x, const float* __restrict__ gam, const float* __restrict__ bet,
    int B, int C, int G, int HW)
{
  __shared__ float rs[256], rss[256], stats[2];
  int bg = blockIdx.x, b = bg / G, g = bg % G;
  int cpg = C / G, n = cpg * HW;
  size_t base = ((size_t)b * C + (size_t)g * cpg) * HW;
  int tid = threadIdx.x;
  float s = 0.0f, ss = 0.0f;
  for (int i = tid; i < n; i += 256) { float v = x[base + i]; s += v; ss += v * v; }
  rs[tid] = s; rss[tid] = ss; __syncthreads();
  for (int o = 128; o > 0; o >>= 1) {
    if (tid < o) { rs[tid] += rs[tid + o]; rss[tid] += rss[tid + o]; }
    __syncthreads();
  }
  if (tid == 0) {
    float m = rs[0] / n, v = rss[0] / n - (rs[0] / n) * (rs[0] / n);
    stats[0] = m; stats[1] = rsqrtf(v + 1e-5f);
  }
  __syncthreads();
  float m = stats[0], r = stats[1];
  for (int i = tid; i < n; i += 256) {
    int c = g * cpg + i / HW;
    x[base + i] = (x[base + i] - m) * r * gam[c] + bet[c];
  }
}

// LayerNorm over last dim (C<=64): one row per thread, in-place capable.
__global__ __launch_bounds__(256) void k_ln(
    const float* __restrict__ in, const float* __restrict__ g, const float* __restrict__ b,
    float* __restrict__ out, int M, int C)
{
  int row = blockIdx.x * blockDim.x + threadIdx.x;
  if (row >= M) return;
  const float* p = in + (size_t)row * C;
  float s = 0.0f;
  for (int c = 0; c < C; ++c) s += p[c];
  float m = s / C, v = 0.0f;
  for (int c = 0; c < C; ++c) { float d = p[c] - m; v += d * d; }
  float r = rsqrtf(v / C + 1e-5f);
  float* o = out + (size_t)row * C;
  for (int c = 0; c < C; ++c) o[c] = (p[c] - m) * r * g[c] + b[c];
}

// maxpool 2x2 then GELU (gelu(max) == reference g(maxpool(.)))
__global__ __launch_bounds__(256) void k_pool_gelu(
    const float* __restrict__ in, float* __restrict__ out, int B, int C, int Ho, int Wo)
{
  int i = blockIdx.x * blockDim.x + threadIdx.x;
  int total = B * C * Ho * Wo;
  if (i >= total) return;
  int x = i % Wo, y = (i / Wo) % Ho, c = (i / (Wo * Ho)) % C, b = i / (Wo * Ho * C);
  int H = 2 * Ho, W = 2 * Wo;
  const float* p = in + ((size_t)(b * C + c) * H + 2 * y) * W + 2 * x;
  float v = fmaxf(fmaxf(p[0], p[1]), fmaxf(p[W], p[W + 1]));
  out[i] = geluf(v);
}

// out = gelu(in) (+ skip)
__global__ __launch_bounds__(256) void k_gelu_add(
    const float* __restrict__ in, const float* __restrict__ skip, float* __restrict__ out, int n)
{
  int i = blockIdx.x * blockDim.x + threadIdx.x;
  if (i >= n) return;
  float v = geluf(in[i]);
  out[i] = skip ? v + skip[i] : v;
}

// bilinear up2 (linspace(0,n-1,2n) grid) + mode: 0 = gelu + skip, 1 = sigmoid
__global__ __launch_bounds__(256) void k_up2(
    const float* __restrict__ in, const float* __restrict__ skip, float* __restrict__ out,
    int B, int C, int H, int W, int mode)
{
  int W2 = 2 * W, H2 = 2 * H;
  int i = blockIdx.x * blockDim.x + threadIdx.x;
  int total = B * C * H2 * W2;
  if (i >= total) return;
  int x2 = i % W2, y2 = (i / W2) % H2, c = (i / (W2 * H2)) % C, b = i / (W2 * H2 * C);
  float cy = (float)y2 * (float)(H - 1) / (float)(2 * H - 1);
  float cx = (float)x2 * (float)(W - 1) / (float)(2 * W - 1);
  int y0 = (int)cy, x0 = (int)cx;
  int y1 = min(y0 + 1, H - 1), x1 = min(x0 + 1, W - 1);
  float wy = cy - y0, wx = cx - x0;
  const float* p = in + (size_t)(b * C + c) * H * W;
  float v = p[y0 * W + x0] * (1 - wy) * (1 - wx) + p[y0 * W + x1] * (1 - wy) * wx
          + p[y1 * W + x0] * wy * (1 - wx)       + p[y1 * W + x1] * wy * wx;
  if (mode == 0)      out[i] = geluf(v) + (skip ? skip[i] : 0.0f);
  else                out[i] = sigmf(v);
}

// ---------------------------------------------------------------------------
// WMMA GEMM path.
//
// Fragment pack: per (tile, kblock) a 16x32 f16 block is stored as 32 lanes x
// 16 contiguous f16 (32 bytes), already in the CDNA5 16-bit A/B VGPR layout:
//   lane = hi*16 + r ;  element j<8 -> K = kb*32 + hi*8 + j
//                       element j>=8 -> K = kb*32 + 16 + hi*8 + (j-8)
// A fragments use row m = tile*16 + r; B fragments use col n = tile*16 + r.
// All M/N/K padding is zero-filled here so the WMMA loop needs no predication.
// ---------------------------------------------------------------------------
__global__ __launch_bounds__(256) void k_packA(
    const float* __restrict__ A, f16_t* __restrict__ out, int M, int K, int Kb, int tM)
{
  int e = blockIdx.x * blockDim.x + threadIdx.x;
  int total = tM * Kb * 512;
  if (e >= total) return;
  int j = e & 15, lane = (e >> 4) & 31, rest = e >> 9;
  int kb = rest % Kb, tile = rest / Kb;
  int r = lane & 15, hi = lane >> 4;
  int m = tile * 16 + r;
  int k = kb * 32 + hi * 8 + ((j < 8) ? j : (8 + j));   // j>=8 -> +16 + (j-8)
  out[e] = (m < M && k < K) ? (f16_t)A[(size_t)m * K + k] : (f16_t)0.0f;
}
__global__ __launch_bounds__(256) void k_packB(
    const float* __restrict__ Bm, f16_t* __restrict__ out, int N, int K, int Kb, int tN)
{
  int e = blockIdx.x * blockDim.x + threadIdx.x;
  int total = tN * Kb * 512;
  if (e >= total) return;
  int j = e & 15, lane = (e >> 4) & 31, rest = e >> 9;
  int kb = rest % Kb, tile = rest / Kb;
  int r = lane & 15, hi = lane >> 4;
  int n = tile * 16 + r;
  int k = kb * 32 + hi * 8 + ((j < 8) ? j : (8 + j));
  out[e] = (n < N && k < K) ? (f16_t)Bm[(size_t)k * N + n] : (f16_t)0.0f;
}

// One wave per 32x16 output: two A fragments share one B fragment ->
// 2 v_wmma per K-block, 3x 32B unpredicated vector loads, f32 accumulate.
// EXEC is all-ones at the WMMAs (only stores are predicated).
__global__ __launch_bounds__(32) void k_gemm_wmma(
    const f16_t* __restrict__ packA, const f16_t* __restrict__ packB,
    const float* __restrict__ bias, float* __restrict__ O,
    int M, int N, int Kb)
{
  int tm0 = blockIdx.x * 2, tm1 = tm0 + 1, tn = blockIdx.y;
  int lane = threadIdx.x;
  int r = lane & 15, hi = lane >> 4;
  const v16h* Av = (const v16h*)packA;
  const v16h* Bv = (const v16h*)packB;
  v8f c0 = {}, c1 = {};
  for (int kb = 0; kb < Kb; ++kb) {
    v16h a0 = Av[((size_t)tm0 * Kb + kb) * 32 + lane];
    v16h a1 = Av[((size_t)tm1 * Kb + kb) * 32 + lane];
    v16h b  = Bv[((size_t)tn  * Kb + kb) * 32 + lane];
    c0 = __builtin_amdgcn_wmma_f32_16x16x32_f16(false, a0, false, b, (short)0, c0, false, false);
    c1 = __builtin_amdgcn_wmma_f32_16x16x32_f16(false, a1, false, b, (short)0, c1, false, false);
  }
  int col = tn * 16 + r;
  float bb = (bias && col < N) ? bias[col] : 0.0f;
#pragma unroll
  for (int j = 0; j < 8; ++j) {                  // C/D layout: vgpr j -> row j + 8*hi
    int row0 = tm0 * 16 + j + 8 * hi;
    int row1 = tm1 * 16 + j + 8 * hi;
    if (row0 < M && col < N) O[(size_t)row0 * N + col] = c0[j] + bb;
    if (row1 < M && col < N) O[(size_t)row1 * N + col] = c1[j] + bb;
  }
}

// ---------------------------------------------------------------------------
// Layout shuffles
// ---------------------------------------------------------------------------
__global__ __launch_bounds__(256) void k_nchw_to_nlc(
    const float* __restrict__ in, float* __restrict__ out, int B, int C, int L)
{
  int i = blockIdx.x * blockDim.x + threadIdx.x;
  if (i >= B * C * L) return;
  int c = i % C, l = (i / C) % L, b = i / (C * L);
  out[i] = in[((size_t)b * C + c) * L + l];
}
__global__ __launch_bounds__(256) void k_nlc_to_nchw(
    const float* __restrict__ in, float* __restrict__ out, int B, int C, int L)
{
  int i = blockIdx.x * blockDim.x + threadIdx.x;
  if (i >= B * C * L) return;
  int l = i % L, c = (i / L) % C, b = i / (L * C);
  out[i] = in[((size_t)b * L + l) * C + c];
}
// xs[(g*B+b)*L+l, d] = xn[(b*L+l)*C + g*dm + d]
__global__ __launch_bounds__(256) void k_build_xs(
    const float* __restrict__ xn, float* __restrict__ xs, int B, int L, int C, int dm)
{
  int i = blockIdx.x * blockDim.x + threadIdx.x;
  if (i >= 4 * B * L * dm) return;
  int d = i % dm, l = (i / dm) % L, q = i / (dm * L);
  int g = q / B, b = q % B;
  xs[i] = xn[((size_t)b * L + l) * C + g * dm + d];
}
// ym[(b*L+l)*C + g*dm+d] = mo[..] + skip*xs[..]
__global__ __launch_bounds__(256) void k_ymc(
    const float* __restrict__ mo, const float* __restrict__ xs,
    const float* __restrict__ skip, float* __restrict__ ym,
    int B, int L, int C, int dm)
{
  int i = blockIdx.x * blockDim.x + threadIdx.x;
  if (i >= B * L * C) return;
  int c = i % C, l = (i / C) % L, b = i / (C * L);
  int g = c / dm, d = c % dm;
  size_t src = ((size_t)(g * B + b) * L + l) * dm + d;
  ym[i] = mo[src] + skip[0] * xs[src];
}

// depthwise conv1d (k=3, SAME) over xs half of xz, then SiLU
__global__ __launch_bounds__(256) void k_dwconv_silu(
    const float* __restrict__ xz, const float* __restrict__ cw,
    const float* __restrict__ cb, float* __restrict__ out, int N, int L, int di)
{
  int i = blockIdx.x * blockDim.x + threadIdx.x;
  if (i >= N * L * di) return;
  int d = i % di, l = (i / di) % L, n = i / (di * L);
  float acc = cb[d];
  for (int k = 0; k < 3; ++k) {
    int ll = l - 1 + k;
    if (ll >= 0 && ll < L) acc += xz[((size_t)n * L + ll) * (2 * di) + d] * cw[d * 3 + k];
  }
  out[i] = siluf(acc);
}

// ---------------------------------------------------------------------------
// Selective scan, atrous step 2.  One block per interleaved sequence
// (q = n*2 + j, element l' -> original l = 2*l'+j).  blockDim = di*16:
// thread t owns state (d = t>>4, nn = t&15); each wave32 holds exactly two
// channels' 16 states, y_d reduced with shfl_xor over 16 lanes.
// dt = softplus(dbl0*dt_w[d] + dt_b[d]) inline (rk == 1 for every block).
// ---------------------------------------------------------------------------
__global__ __launch_bounds__(512) void k_scan(
    const float* __restrict__ xsc, const float* __restrict__ dbl,
    const float* __restrict__ A_log, const float* __restrict__ Dp,
    const float* __restrict__ dt_w, const float* __restrict__ dt_b,
    float* __restrict__ y, int L, int di)
{
  int q = blockIdx.x, n = q >> 1, j = q & 1;
  int t = threadIdx.x, d = t >> 4, nn = t & 15;
  float An  = -__expf(A_log[d * 16 + nn]);
  float dtw = dt_w[d], dtb = dt_b[d], Dd = Dp[d];
  float h = 0.0f;
  int Lh = L >> 1;
  for (int lp = 0; lp < Lh; ++lp) {
    size_t idx = (size_t)n * L + (lp * 2 + j);
    const float* dr = dbl + idx * 33;
    float dt = softplusf_(dr[0] * dtw + dtb);
    float u  = xsc[idx * di + d];
    h = __expf(dt * An) * h + dt * dr[1 + nn] * u;
    float p = h * dr[17 + nn];
    p += __shfl_xor(p, 1, 32);
    p += __shfl_xor(p, 2, 32);
    p += __shfl_xor(p, 4, 32);
    p += __shfl_xor(p, 8, 32);
    if (nn == 0) y[idx * di + d] = p + u * Dd;
  }
}

// y *= silu(z), z = xz[:, di:2di]
__global__ __launch_bounds__(256) void k_gate(
    float* __restrict__ y, const float* __restrict__ xz, int rows, int di)
{
  int i = blockIdx.x * blockDim.x + threadIdx.x;
  if (i >= rows * di) return;
  int d = i % di, row = i / di;
  y[i] *= siluf(xz[(size_t)row * 2 * di + di + d]);
}

// ---------------------------------------------------------------------------
// Bridge kernels
// ---------------------------------------------------------------------------
__global__ __launch_bounds__(256) void k_meanmax(
    const float* __restrict__ t, float* __restrict__ a2, int B, int C, int HW)
{
  int i = blockIdx.x * blockDim.x + threadIdx.x;
  if (i >= B * HW) return;
  int p = i % HW, b = i / HW;
  float s = 0.0f, mx = -3.4e38f;
  for (int c = 0; c < C; ++c) {
    float v = t[((size_t)b * C + c) * HW + p];
    s += v; mx = fmaxf(mx, v);
  }
  a2[((size_t)b * 2 + 0) * HW + p] = s / C;
  a2[((size_t)b * 2 + 1) * HW + p] = mx;
}
__global__ __launch_bounds__(256) void k_satt_combine(
    const float* __restrict__ t, const float* __restrict__ psatt,
    float* __restrict__ ts1, float* __restrict__ ts2, int B, int C, int HW)
{
  int i = blockIdx.x * blockDim.x + threadIdx.x;
  if (i >= B * C * HW) return;
  int p = i % HW, b = i / (HW * C);
  float s = sigmf(psatt[(size_t)b * HW + p]);
  float v1 = s * t[i];
  ts1[i] = v1;
  ts2[i] = v1 + t[i];
}
__global__ __launch_bounds__(256) void k_chanmean(
    const float* __restrict__ ts2, float* __restrict__ attin, int B, int C, int HW, int off)
{
  __shared__ float red[256];
  int bc = blockIdx.x, b = bc / C, c = bc % C;
  size_t base = ((size_t)b * C + c) * HW;
  int tid = threadIdx.x;
  float s = 0.0f;
  for (int i = tid; i < HW; i += 256) s += ts2[base + i];
  red[tid] = s; __syncthreads();
  for (int o = 128; o > 0; o >>= 1) { if (tid < o) red[tid] += red[tid + o]; __syncthreads(); }
  if (tid == 0) attin[b * 128 + off + c] = red[0] / HW;
}
__global__ __launch_bounds__(256) void k_catt_conv1d(
    const float* __restrict__ attin, const float* __restrict__ w, float* __restrict__ attout, int B)
{
  int i = blockIdx.x * blockDim.x + threadIdx.x;
  if (i >= B * 128) return;
  int jx = i % 128, b = i / 128;
  float acc = 0.0f;
  for (int k = 0; k < 3; ++k) {
    int jj = jx - 1 + k;
    if (jj >= 0 && jj < 128) acc += attin[b * 128 + jj] * w[k];
  }
  attout[i] = acc;
}
__global__ __launch_bounds__(256) void k_catt_ai(
    const float* __restrict__ attout, const float* __restrict__ w,
    const float* __restrict__ bias, float* __restrict__ ai, int B, int Ci)
{
  int i = blockIdx.x * blockDim.x + threadIdx.x;
  if (i >= B * Ci) return;
  int c = i % Ci, b = i / Ci;
  float acc = bias[c];
  for (int jx = 0; jx < 128; ++jx) acc += attout[b * 128 + jx] * w[jx * Ci + c];
  ai[i] = sigmf(acc);
}
// ts1 <- ai*ts2 + ts1 (in-place; ts1 becomes the bridge output)
__global__ __launch_bounds__(256) void k_catt_combine(
    float* __restrict__ ts1, const float* __restrict__ ts2,
    const float* __restrict__ ai, int B, int C, int HW)
{
  int i = blockIdx.x * blockDim.x + threadIdx.x;
  if (i >= B * C * HW) return;
  int c = (i / HW) % C, b = i / (HW * C);
  ts1[i] = ai[b * C + c] * ts2[i] + ts1[i];
}

// ---------------------------------------------------------------------------
// Host side
// ---------------------------------------------------------------------------
static inline int cdiv(int a, int b) { return (a + b - 1) / b; }

struct LinP { const float *b, *w; };
struct GnP  { const float *b, *g; };
struct PvmP {
  const float *ln_b, *ln_g;
  const float *A_log, *D, *conv_b, *conv_w, *dt_b, *dt_w;
  const float *in_b, *in_w, *out_b, *out_w, *xproj_w;
  const float *proj_b, *proj_w, *skip;
};

struct Scratch {
  float *xf, *xs, *xz, *xsc, *dbl, *y, *mo, *ym, *po;
  f16_t *pa, *pb;
};

static void gemm_run(hipStream_t s, const float* A, const float* Bw, const float* bias,
                     float* O, int M, int N, int K, const Scratch& sc)
{
  int Kb = cdiv(K, 32);
  int tM = cdiv(M, 32) * 2;      // even tile count (zero-padded by pack)
  int tN = cdiv(N, 16);
  k_packA<<<cdiv(tM * Kb * 512, 256), 256, 0, s>>>(A, sc.pa, M, K, Kb, tM);
  k_packB<<<cdiv(tN * Kb * 512, 256), 256, 0, s>>>(Bw, sc.pb, N, K, Kb, tN);
  dim3 g(tM / 2, tN);
  k_gemm_wmma<<<g, 32, 0, s>>>(sc.pa, sc.pb, bias, O, M, N, Kb);
}

static void pvm_run(hipStream_t s, const float* inNCHW, float* outNCHW,
                    const PvmP& p, int B, int C, int H, int W, int co, const Scratch& sc)
{
  int L = H * W, dm = C / 4, di = 2 * dm, N = 4 * B;
  int ML = B * L, NL = N * L;
  k_nchw_to_nlc<<<cdiv(ML * C, 256), 256, 0, s>>>(inNCHW, sc.xf, B, C, L);
  k_ln<<<cdiv(ML, 256), 256, 0, s>>>(sc.xf, p.ln_g, p.ln_b, sc.xf, ML, C);
  k_build_xs<<<cdiv(NL * dm, 256), 256, 0, s>>>(sc.xf, sc.xs, B, L, C, dm);
  gemm_run(s, sc.xs, p.in_w, p.in_b, sc.xz, NL, 2 * di, dm, sc);
  k_dwconv_silu<<<cdiv(NL * di, 256), 256, 0, s>>>(sc.xz, p.conv_w, p.conv_b, sc.xsc, N, L, di);
  gemm_run(s, sc.xsc, p.xproj_w, nullptr, sc.dbl, NL, 33, di, sc);
  k_scan<<<N * 2, di * 16, 0, s>>>(sc.xsc, sc.dbl, p.A_log, p.D, p.dt_w, p.dt_b, sc.y, L, di);
  k_gate<<<cdiv(NL * di, 256), 256, 0, s>>>(sc.y, sc.xz, NL, di);
  gemm_run(s, sc.y, p.out_w, p.out_b, sc.mo, NL, dm, di, sc);
  k_ymc<<<cdiv(ML * C, 256), 256, 0, s>>>(sc.mo, sc.xs, p.skip, sc.ym, B, L, C, dm);
  k_ln<<<cdiv(ML, 256), 256, 0, s>>>(sc.ym, p.ln_g, p.ln_b, sc.ym, ML, C);
  gemm_run(s, sc.ym, p.proj_w, p.proj_b, sc.po, ML, co, C, sc);
  k_nlc_to_nchw<<<cdiv(ML * co, 256), 256, 0, s>>>(sc.po, outNCHW, B, co, L);
}

extern "C" void kernel_launch(void* const* d_in, const int* in_sizes, int n_in,
                              void* d_out, int out_size, void* d_ws, size_t ws_size,
                              hipStream_t stream)
{
  (void)in_sizes; (void)out_size; (void)ws_size;
  if (n_in < 142) return;
  int ix = 0;
  auto F = [&](void) { return (const float*)d_in[ix++]; };

  // ---- unpack (jax pytree order: dict keys sorted; x first per setup_inputs)
  const float* X = F();
  LinP catt[5];
  for (int i = 0; i < 5; ++i) { catt[i].b = F(); catt[i].w = F(); }
  const float* catt_conv = F();
  GnP dbn[5];
  for (int i = 0; i < 5; ++i) { dbn[i].b = F(); dbn[i].g = F(); }
  auto readPVM = [&](PvmP& p) {
    p.ln_b = F(); p.ln_g = F();
    p.A_log = F(); p.D = F(); p.conv_b = F(); p.conv_w = F();
    p.dt_b = F(); p.dt_w = F(); p.in_b = F(); p.in_w = F();
    p.out_b = F(); p.out_w = F(); p.xproj_w = F();
    p.proj_b = F(); p.proj_w = F(); p.skip = F();
  };
  PvmP dec1, dec2, dec3; readPVM(dec1); readPVM(dec2); readPVM(dec3);
  LinP dec4, dec5; dec4.b = F(); dec4.w = F(); dec5.b = F(); dec5.w = F();
  GnP ebn[5];
  for (int i = 0; i < 5; ++i) { ebn[i].b = F(); ebn[i].g = F(); }
  LinP enc1, enc2, enc3;
  enc1.b = F(); enc1.w = F(); enc2.b = F(); enc2.w = F(); enc3.b = F(); enc3.w = F();
  PvmP enc4, enc5, enc6; readPVM(enc4); readPVM(enc5); readPVM(enc6);
  LinP finalc, satt;
  finalc.b = F(); finalc.w = F(); satt.b = F(); satt.w = F();

  // ---- workspace bump allocator
  char* base = (char*)d_ws;
  size_t off = 0;
  auto alloc = [&](size_t nf) {
    float* p = (float*)(base + off);
    off = (off + nf * sizeof(float) + 255) & ~(size_t)255;
    return p;
  };
  const int B = 8;
  float* bigA = alloc(16777216);              // largest conv scratch (8,8,512,512)
  float* t1 = alloc(4194304);                 // (8,8,256,256)
  float* t2 = alloc(2097152);                 // (8,16,128,128)
  float* t3 = alloc(786432);                  // (8,24,64,64)
  float* t4 = alloc(262144);                  // (8,32,32,32)
  float* t5 = alloc(98304);                   // (8,48,16,16)
  Scratch sc;
  sc.xf  = alloc(786432);  sc.xs = alloc(786432);  sc.xz = alloc(3145728);
  sc.xsc = alloc(1572864); sc.dbl = alloc(4325376); sc.y = alloc(1572864);
  sc.mo  = alloc(786432);  sc.ym = alloc(786432);  sc.po = alloc(1048576);
  sc.pa  = (f16_t*)alloc(2621440);            // 5.24M f16 fragment buffer (A)
  sc.pb  = (f16_t*)alloc(16384);              // 32K f16 fragment buffer (B)
  float* pvout = alloc(1048576);
  // bridge
  const int lvC[5]  = {8, 16, 24, 32, 48};
  const int lvH[5]  = {256, 128, 64, 32, 16};
  const int lvOff[5]= {0, 8, 24, 48, 80};
  float* tlev[5] = {t1, t2, t3, t4, t5};
  float* ts1b[5]; float* ts2b[5];
  for (int i = 0; i < 5; ++i) {
    size_t sz = (size_t)B * lvC[i] * lvH[i] * lvH[i];
    ts1b[i] = alloc(sz); ts2b[i] = alloc(sz);
  }
  float* a2    = alloc(1048576);
  float* psatt = alloc(524288);
  float* attin = alloc(1024);
  float* attout= alloc(1024);
  float* aibuf = alloc(B * 48);
  float* u6 = alloc(131072);
  float* o5 = alloc(98304);
  float* o4 = alloc(262144);
  float* o3 = alloc(786432);
  float* o2 = alloc(2097152);
  float* o1 = alloc(4194304);

  hipStream_t s = stream;
  // ================= encoder =================
  k_conv2d<<<cdiv(B*8*512*512,256),256,0,s>>>(X, enc1.w, enc1.b, bigA, B,3,8,512,512,3,1,1);
  k_gn<<<B*4,256,0,s>>>(bigA, ebn[0].g, ebn[0].b, B, 8, 4, 512*512);
  k_pool_gelu<<<cdiv(B*8*256*256,256),256,0,s>>>(bigA, t1, B, 8, 256, 256);
  k_conv2d<<<cdiv(B*16*256*256,256),256,0,s>>>(t1, enc2.w, enc2.b, bigA, B,8,16,256,256,3,1,1);
  k_gn<<<B*4,256,0,s>>>(bigA, ebn[1].g, ebn[1].b, B, 16, 4, 256*256);
  k_pool_gelu<<<cdiv(B*16*128*128,256),256,0,s>>>(bigA, t2, B, 16, 128, 128);
  k_conv2d<<<cdiv(B*24*128*128,256),256,0,s>>>(t2, enc3.w, enc3.b, bigA, B,16,24,128,128,3,1,1);
  k_gn<<<B*4,256,0,s>>>(bigA, ebn[2].g, ebn[2].b, B, 24, 4, 128*128);
  k_pool_gelu<<<cdiv(B*24*64*64,256),256,0,s>>>(bigA, t3, B, 24, 64, 64);
  pvm_run(s, t3, pvout, enc4, B, 24, 64, 64, 32, sc);
  k_gn<<<B*4,256,0,s>>>(pvout, ebn[3].g, ebn[3].b, B, 32, 4, 64*64);
  k_pool_gelu<<<cdiv(B*32*32*32,256),256,0,s>>>(pvout, t4, B, 32, 32, 32);
  pvm_run(s, t4, pvout, enc5, B, 32, 32, 32, 48, sc);
  k_gn<<<B*4,256,0,s>>>(pvout, ebn[4].g, ebn[4].b, B, 48, 4, 32*32);
  k_pool_gelu<<<cdiv(B*48*16*16,256),256,0,s>>>(pvout, t5, B, 48, 16, 16);

  // ================= bridge =================
  for (int i = 0; i < 5; ++i) {
    int C = lvC[i], H = lvH[i], HW = H * H;
    k_meanmax<<<cdiv(B*HW,256),256,0,s>>>(tlev[i], a2, B, C, HW);
    k_conv2d<<<cdiv(B*1*HW,256),256,0,s>>>(a2, satt.w, satt.b, psatt, B,2,1,H,H,7,9,3);
    k_satt_combine<<<cdiv(B*C*HW,256),256,0,s>>>(tlev[i], psatt, ts1b[i], ts2b[i], B, C, HW);
    k_chanmean<<<B*C,256,0,s>>>(ts2b[i], attin, B, C, HW, lvOff[i]);
  }
  k_catt_conv1d<<<cdiv(B*128,256),256,0,s>>>(attin, catt_conv, attout, B);
  for (int i = 0; i < 5; ++i) {
    int C = lvC[i], HW = lvH[i] * lvH[i];
    k_catt_ai<<<cdiv(B*C,256),256,0,s>>>(attout, catt[i].w, catt[i].b, aibuf, B, C);
    k_catt_combine<<<cdiv(B*C*HW,256),256,0,s>>>(ts1b[i], ts2b[i], aibuf, B, C, HW);
  }
  // bridge outputs now live in ts1b[i]; original t5 preserved for enc6 input

  // ================= decoder =================
  pvm_run(s, t5, pvout, enc6, B, 48, 16, 16, 64, sc);
  k_gelu_add<<<cdiv(B*64*256,256),256,0,s>>>(pvout, nullptr, u6, B*64*256);
  pvm_run(s, u6, pvout, dec1, B, 64, 16, 16, 48, sc);
  k_gn<<<B*4,256,0,s>>>(pvout, dbn[0].g, dbn[0].b, B, 48, 4, 256);
  k_gelu_add<<<cdiv(B*48*256,256),256,0,s>>>(pvout, ts1b[4], o5, B*48*256);
  pvm_run(s, o5, pvout, dec2, B, 48, 16, 16, 32, sc);
  k_gn<<<B*4,256,0,s>>>(pvout, dbn[1].g, dbn[1].b, B, 32, 4, 256);
  k_up2<<<cdiv(B*32*32*32,256),256,0,s>>>(pvout, ts1b[3], o4, B, 32, 16, 16, 0);
  pvm_run(s, o4, pvout, dec3, B, 32, 32, 32, 24, sc);
  k_gn<<<B*4,256,0,s>>>(pvout, dbn[2].g, dbn[2].b, B, 24, 4, 1024);
  k_up2<<<cdiv(B*24*64*64,256),256,0,s>>>(pvout, ts1b[2], o3, B, 24, 32, 32, 0);
  k_conv2d<<<cdiv(B*16*64*64,256),256,0,s>>>(o3, dec4.w, dec4.b, bigA, B,24,16,64,64,3,1,1);
  k_gn<<<B*4,256,0,s>>>(bigA, dbn[3].g, dbn[3].b, B, 16, 4, 64*64);
  k_up2<<<cdiv(B*16*128*128,256),256,0,s>>>(bigA, ts1b[1], o2, B, 16, 64, 64, 0);
  k_conv2d<<<cdiv(B*8*128*128,256),256,0,s>>>(o2, dec5.w, dec5.b, bigA, B,16,8,128,128,3,1,1);
  k_gn<<<B*4,256,0,s>>>(bigA, dbn[4].g, dbn[4].b, B, 8, 4, 128*128);
  k_up2<<<cdiv(B*8*256*256,256),256,0,s>>>(bigA, ts1b[0], o1, B, 8, 128, 128, 0);
  k_conv2d<<<cdiv(B*1*256*256,256),256,0,s>>>(o1, finalc.w, finalc.b, bigA, B,8,1,256,256,3,1,1);
  k_up2<<<cdiv(B*1*512*512,256),256,0,s>>>(bigA, nullptr, (float*)d_out, B, 1, 256, 256, 1);
}
